// MultiheadAttention_22187801051599
// MI455X (gfx1250) — compile-verified, matrix operand order checked by reading, not run on hardware
//
#include <hip/hip_runtime.h>
#include <hip/hip_bf16.h>

// CDNA5 (gfx1250) wave32 WMMA types
typedef __attribute__((ext_vector_type(16))) __bf16 v16bf;
typedef __attribute__((ext_vector_type(8)))  __bf16 v8bf;
typedef __attribute__((ext_vector_type(4)))  __bf16 v4bf;
typedef __attribute__((ext_vector_type(8)))  float  v8f;

#define B_  4
#define S_  1024
#define D_  1024
#define H_  16
#define HD_ 64
#define N3D 3072
#define HEAD_ELEMS ((size_t)B_ * H_ * S_ * HD_)   // elems per Q/K/V tensor in ws

__device__ __forceinline__ v16bf join8(v8bf lo, v8bf hi) {
  return __builtin_shufflevector(lo, hi, 0,1,2,3,4,5,6,7,8,9,10,11,12,13,14,15);
}

// Async copy of 16 bytes/lane from global to LDS (GLOBAL_LOAD_ASYNC_TO_LDS_B128,
// tracked on ASYNCcnt). ldsoff = wave-relative LDS byte offset (= low 32 bits of
// the generic pointer per the LDS aperture mapping), gaddr = 64-bit global addr.
__device__ __forceinline__ void async_copy_b128(const void* lds_ptr, const void* gptr) {
  unsigned ldsoff = (unsigned)(uintptr_t)lds_ptr;
  unsigned long long gaddr = (unsigned long long)(uintptr_t)gptr;
  asm volatile("global_load_async_to_lds_b128 %0, %1, off"
               :: "v"(ldsoff), "v"(gaddr) : "memory");
}

// ---------------------------------------------------------------------------
// Phase 1: qkv = x @ W + b   (M=4096, K=1024, N=3072), bf16 WMMA, f32 accum.
// Workspace layout: Q [b][h][s][d], K [b][h][s][d], V pre-transposed [b][h][d][s].
// Block: 256 thr (8 waves). Block tile 128(M) x 128(N); wave tile 64x32;
// k staged 64 per barrier, consumed in two 32-deep WMMA passes.
// ---------------------------------------------------------------------------
__global__ __launch_bounds__(256)
void qkv_proj_kernel(const float* __restrict__ x, const float* __restrict__ W,
                     const float* __restrict__ bias, __bf16* __restrict__ ws)
{
  __shared__ __bf16 As[128 * 72];   // [row][k], stride 72 halves (144B rows, 16B-aligned)
  __shared__ __bf16 BTs[128 * 72];  // [col][k] (W transposed), stride 72

  const int tid  = threadIdx.x;
  const int lane = tid & 31;
  const int wave = tid >> 5;       // 0..7
  const int lm   = lane & 15;
  const int hs   = lane >> 4;

  const int blockM = blockIdx.x & 31;   // 4096/128 = 32
  const int blockN = blockIdx.x >> 5;   // 3072/128 = 24
  const int R0 = blockM * 128;
  const int C0 = blockN * 128;

  const int wm = wave & 1;   // 0..1 -> 64-row sub-tile
  const int wn = wave >> 1;  // 0..3 -> 32-col sub-tile

  v8f acc[4][2] = {};

  for (int k0 = 0; k0 < D_; k0 += 64) {
    __syncthreads();
    // Batched staging: issue all 16 global float4 loads first (load clause),
    // then convert f32->bf16 and store to LDS.
    float4 va[8], vb[8];
    #pragma unroll
    for (int i = 0; i < 8; ++i) {
      int idx = i * 256 + tid;             // A: 2048 float4 segments
      int r = idx >> 4, c4 = (idx & 15) * 4;
      va[i] = *(const float4*)(x + (size_t)(R0 + r) * D_ + k0 + c4);
    }
    #pragma unroll
    for (int i = 0; i < 8; ++i) {
      int idx = i * 256 + tid;             // W: 2048 float4 segments
      int kr = idx >> 5, c4 = (idx & 31) * 4;
      vb[i] = *(const float4*)(W + (size_t)(k0 + kr) * N3D + C0 + c4);
    }
    #pragma unroll
    for (int i = 0; i < 8; ++i) {
      int idx = i * 256 + tid;
      int r = idx >> 4, c4 = (idx & 15) * 4;
      v4bf b; b[0] = (__bf16)va[i].x; b[1] = (__bf16)va[i].y;
              b[2] = (__bf16)va[i].z; b[3] = (__bf16)va[i].w;
      *(v4bf*)&As[r * 72 + c4] = b;
    }
    #pragma unroll
    for (int i = 0; i < 8; ++i) {
      int idx = i * 256 + tid;
      int kr = idx >> 5, c4 = (idx & 31) * 4;
      BTs[(c4 + 0) * 72 + kr] = (__bf16)vb[i].x;
      BTs[(c4 + 1) * 72 + kr] = (__bf16)vb[i].y;
      BTs[(c4 + 2) * 72 + kr] = (__bf16)vb[i].z;
      BTs[(c4 + 3) * 72 + kr] = (__bf16)vb[i].w;
    }
    // Prefetch next k-panel while this one is consumed (global_prefetch_b8)
    if (k0 + 64 < D_) {
      __builtin_prefetch(x + (size_t)(R0 + (tid >> 4)) * D_ + k0 + 64 + (tid & 15) * 4, 0, 3);
      __builtin_prefetch(W + (size_t)(k0 + 64 + (tid >> 5)) * N3D + C0 + (tid & 31) * 4, 0, 3);
    }
    __syncthreads();

    // Two 32-deep WMMA passes over the staged 64-k panel
    #pragma unroll
    for (int ks = 0; ks < 64; ks += 32) {
      v16bf afrag[4], bfrag[2];
      #pragma unroll
      for (int mt = 0; mt < 4; ++mt) {
        const __bf16* p = &As[(wm * 64 + mt * 16 + lm) * 72 + ks];
        afrag[mt] = join8(*(const v8bf*)(p + hs * 8), *(const v8bf*)(p + 16 + hs * 8));
      }
      #pragma unroll
      for (int nt = 0; nt < 2; ++nt) {
        const __bf16* p = &BTs[(wn * 32 + nt * 16 + lm) * 72 + ks];
        bfrag[nt] = join8(*(const v8bf*)(p + hs * 8), *(const v8bf*)(p + 16 + hs * 8));
      }
      #pragma unroll
      for (int mt = 0; mt < 4; ++mt)
        #pragma unroll
        for (int nt = 0; nt < 2; ++nt)
          acc[mt][nt] = __builtin_amdgcn_wmma_f32_16x16x32_bf16(
              false, afrag[mt], false, bfrag[nt], (short)0, acc[mt][nt], false, false);
    }
  }

  // Epilogue: + bias, scatter bf16 into ws (V transposed)
  #pragma unroll
  for (int nt = 0; nt < 2; ++nt) {
    const int c  = C0 + wn * 32 + nt * 16 + lm;      // column in [0,3072)
    const float bv = bias[c];
    const int h   = c / 192;
    const int rem = c - h * 192;
    const int t   = rem / 64;          // 0=q 1=k 2=v
    const int d   = rem - t * 64;
    __bf16* basep = ws + (size_t)t * HEAD_ELEMS;
    #pragma unroll
    for (int mt = 0; mt < 4; ++mt) {
      #pragma unroll
      for (int r = 0; r < 8; ++r) {
        const int m = R0 + wm * 64 + mt * 16 + r + 8 * hs;  // C-layout row
        const int b = m >> 10;
        const int s = m & 1023;
        const size_t idx = (t == 2)
            ? (((size_t)(b * H_ + h)) * HD_ + d) * S_ + s    // V: [b][h][d][s]
            : (((size_t)(b * H_ + h)) * S_ + s) * HD_ + d;   // Q,K: [b][h][s][d]
        basep[idx] = (__bf16)(acc[mt][nt][r] + bv);
      }
    }
  }
}

// ---------------------------------------------------------------------------
// Phase 2: flash-style attention. One block per (b,h,128-q slab).
// 8 waves x 16 q rows; online softmax over key chunks of 64.
// K/V staged into LDS with GLOBAL_LOAD_ASYNC_TO_LDS_B128 (ASYNCcnt),
// V arrives pre-transposed from phase 1 -> contiguous copies, no transpose.
// ---------------------------------------------------------------------------
__global__ __launch_bounds__(256)
void attn_kernel(const __bf16* __restrict__ ws, float* __restrict__ out)
{
  __shared__ __bf16 Kc[64 * 72];        // [key][d], stride 72 halves
  __shared__ __bf16 VcT[64 * 72];       // [d][key], stride 72
  __shared__ __bf16 pbuf[8][16 * 32];   // per-wave P tile, row-major 16x32

  const int tid  = threadIdx.x;
  const int lane = tid & 31;
  const int wave = tid >> 5;
  const int lm   = lane & 15;
  const int hs   = lane >> 4;

  const int bh   = blockIdx.x >> 3;   // 0..63
  const int qblk = blockIdx.x & 7;    // 0..7
  const int b    = bh >> 4;
  const int h    = bh & 15;
  const int q0   = qblk * 128 + wave * 16;   // this wave's 16 q rows

  const __bf16* Qg = ws + (((size_t)(b * H_ + h)) * S_) * HD_;
  const __bf16* Kg = Qg + HEAD_ELEMS;
  const __bf16* Vt = ws + 2 * HEAD_ELEMS + (((size_t)(b * H_ + h)) * HD_) * S_; // [d][s]

  // Q fragments (d 0..31 and 32..63), held in registers for the whole kernel
  v16bf qfrag[2];
  {
    const __bf16* qrow = Qg + (size_t)(q0 + lm) * HD_;
    #pragma unroll
    for (int df = 0; df < 2; ++df)
      qfrag[df] = join8(*(const v8bf*)(qrow + df * 32 + hs * 8),
                        *(const v8bf*)(qrow + df * 32 + 16 + hs * 8));
  }

  float mrow[8], lrow[8];
  v8f o[4] = {};
  #pragma unroll
  for (int r = 0; r < 8; ++r) { mrow[r] = -3.0e38f; lrow[r] = 0.f; }

  for (int kc = 0; kc < S_; kc += 64) {
    __syncthreads();   // previous chunk fully consumed before overwrite
    // Async-stage K [key][d] and V [d][key]: 512 x 16B segments each
    #pragma unroll
    for (int i = 0; i < 2; ++i) {
      int idx = i * 256 + tid;
      int r = idx >> 3;                  // row 0..63 (key for K, d for V)
      int c = (idx & 7) * 8;             // col base (halves)
      async_copy_b128(&Kc[r * 72 + c],  Kg + (size_t)(kc + r) * HD_ + c);
      async_copy_b128(&VcT[r * 72 + c], Vt + (size_t)r * S_ + kc + c);
    }
    // Prefetch next key chunk into cache while computing this one
    if (kc + 64 < S_) {
      __builtin_prefetch(Kg + (size_t)(kc + 64 + (tid >> 3)) * HD_ + (tid & 7) * 8, 0, 3);
      __builtin_prefetch(Vt + (size_t)(tid >> 3) * S_ + kc + 64 + (tid & 7) * 8, 0, 3);
    }
    asm volatile("s_wait_asynccnt 0" ::: "memory");  // this wave's async copies done
    __syncthreads();                                  // ... and everyone else's

    #pragma unroll
    for (int sub = 0; sub < 2; ++sub) {
      const int kb = sub * 32;

      // scores: 16 q x 32 keys (two 16x16 tiles), K-dim = 64 split over 2 WMMAs
      v8f sc[2];
      #pragma unroll
      for (int jj = 0; jj < 2; ++jj) {
        v8f cacc = {};
        const __bf16* krow = &Kc[(kb + jj * 16 + lm) * 72];
        v16bf kf0 = join8(*(const v8bf*)(krow + hs * 8),
                          *(const v8bf*)(krow + 16 + hs * 8));
        v16bf kf1 = join8(*(const v8bf*)(krow + 32 + hs * 8),
                          *(const v8bf*)(krow + 48 + hs * 8));
        cacc = __builtin_amdgcn_wmma_f32_16x16x32_bf16(false, qfrag[0], false, kf0,
                                                       (short)0, cacc, false, false);
        cacc = __builtin_amdgcn_wmma_f32_16x16x32_bf16(false, qfrag[1], false, kf1,
                                                       (short)0, cacc, false, false);
        sc[jj] = cacc * 0.125f;          // 1/sqrt(64)
      }

      // online softmax: row max / sum via shfl-xor within 16-lane halves
      float mnew[8], alpha[8], rs[8];
      #pragma unroll
      for (int r = 0; r < 8; ++r) {
        float v = fmaxf(sc[0][r], sc[1][r]);
        #pragma unroll
        for (int off = 1; off < 16; off <<= 1)
          v = fmaxf(v, __shfl_xor(v, off, 32));
        mnew[r]  = fmaxf(mrow[r], v);
        alpha[r] = __expf(mrow[r] - mnew[r]);
      }
      #pragma unroll
      for (int r = 0; r < 8; ++r) {
        float p0 = __expf(sc[0][r] - mnew[r]);
        float p1 = __expf(sc[1][r] - mnew[r]);
        sc[0][r] = p0; sc[1][r] = p1;
        float v = p0 + p1;
        #pragma unroll
        for (int off = 1; off < 16; off <<= 1)
          v += __shfl_xor(v, off, 32);
        rs[r] = v;
      }
      #pragma unroll
      for (int r = 0; r < 8; ++r) {
        lrow[r] = lrow[r] * alpha[r] + rs[r];
        mrow[r] = mnew[r];
      }
      #pragma unroll
      for (int dt = 0; dt < 4; ++dt)
        #pragma unroll
        for (int r = 0; r < 8; ++r)
          o[dt][r] *= alpha[r];

      // C-layout -> A-layout via wave-private LDS round trip
      __bf16* pb = &pbuf[wave][0];
      #pragma unroll
      for (int jj = 0; jj < 2; ++jj)
        #pragma unroll
        for (int r = 0; r < 8; ++r)
          pb[(r + 8 * hs) * 32 + jj * 16 + lm] = (__bf16)sc[jj][r];
      asm volatile("s_wait_dscnt 0" ::: "memory");   // DS in-order per wave; force visibility

      const __bf16* prow = pb + lm * 32;
      v16bf pfrag = join8(*(const v8bf*)(prow + hs * 8),
                          *(const v8bf*)(prow + 16 + hs * 8));

      // O += P @ V  (4 d-tiles of 16); V B-frag rows are contiguous in VcT
      #pragma unroll
      for (int dt = 0; dt < 4; ++dt) {
        const __bf16* vrow = &VcT[(dt * 16 + lm) * 72 + kb];
        v16bf vfrag = join8(*(const v8bf*)(vrow + hs * 8),
                            *(const v8bf*)(vrow + 16 + hs * 8));
        o[dt] = __builtin_amdgcn_wmma_f32_16x16x32_bf16(false, pfrag, false, vfrag,
                                                        (short)0, o[dt], false, false);
      }
    }
  }

  // normalize and write out[b, q, h*64 + d] (f32)
  #pragma unroll
  for (int dt = 0; dt < 4; ++dt) {
    const int d = dt * 16 + lm;
    #pragma unroll
    for (int r = 0; r < 8; ++r) {
      const int q = q0 + r + 8 * hs;
      out[((size_t)(b * S_ + q)) * D_ + h * HD_ + d] = o[dt][r] / lrow[r];
    }
  }
}

// ---------------------------------------------------------------------------
extern "C" void kernel_launch(void* const* d_in, const int* in_sizes, int n_in,
                              void* d_out, int out_size, void* d_ws, size_t ws_size,
                              hipStream_t stream) {
  const float* x    = (const float*)d_in[0];   // [B,S,D] f32
  const float* W    = (const float*)d_in[1];   // [D,3D]  f32
  const float* bias = (const float*)d_in[2];   // [3D]    f32
  float* out        = (float*)d_out;           // [B,S,D] f32
  __bf16* ws        = (__bf16*)d_ws;           // bf16 QKV staging, ~24 MB

  qkv_proj_kernel<<<dim3(32 * 24), 256, 0, stream>>>(x, W, bias, ws);
  attn_kernel<<<dim3(B_ * H_ * (S_ / 128)), 256, 0, stream>>>(ws, out);
}
